// RTDetrDecoderLayer_47287589929536
// MI455X (gfx1250) — matrix operand order repaired and checked by reading, not compiled, not run on hardware
//
#include <hip/hip_runtime.h>
#include <hip/hip_bf16.h>

// ---------------------------------------------------------------------------
// RT-DETR decoder layer for gfx1250 (MI455X).
// All GEMMs: bf16 operands, v_wmma_f32_16x16x32_bf16, double-buffered LDS
// tiles filled with global_load_async_to_lds_b128 (ASYNCcnt path).
// B=32 NQ=300 D=256 H=8 HD=32 L=3 P=4 FFN=1024 S=8400
// ---------------------------------------------------------------------------

typedef __attribute__((ext_vector_type(16))) __bf16 v16bf;
typedef __attribute__((ext_vector_type(8)))  float  v8f;

union FragBF { v16bf v; unsigned int u[8]; };

__device__ __forceinline__ unsigned short f2bf(float f) {
    union { float f; unsigned int u; } x; x.f = f;
    unsigned int u = x.u;
    unsigned int r = u + 0x7FFFu + ((u >> 16) & 1u);   // round-to-nearest-even
    return (unsigned short)(r >> 16);
}
__device__ __forceinline__ float bf2f(unsigned short u) {
    union { unsigned int u; float f; } x; x.u = ((unsigned int)u) << 16;
    return x.f;
}

// A-matrix (16x32, 16-bit) pair base K for vgpr j, lane-half h (ISA 7.12.2)
__device__ __forceinline__ int kpairA(int j, int half) {
    return ((j >> 2) << 4) + ((j & 3) << 1) + (half ? 8 : 0);
}
// B-matrix (32x16, 16-bit) pair base K for vgpr j, lane-half h (same K order as A)
__device__ __forceinline__ int kpairB(int j, int half) {
    return ((j >> 2) << 4) + ((j & 3) << 1) + (half ? 8 : 0);
}

// LDS (addrspace 3) byte offset of a generic pointer into __shared__
typedef __attribute__((address_space(3))) const unsigned short lds_cus;
__device__ __forceinline__ unsigned lds_off(const unsigned short* p) {
    return (unsigned)(unsigned long long)(lds_cus*)p;
}

// async global -> LDS, 16B per lane, GVS addressing (saddr + u32 voffset)
__device__ __forceinline__ void async_b128(unsigned lds, unsigned long long gbase, unsigned voff) {
    asm volatile("global_load_async_to_lds_b128 %0, %1, %2"
                 :: "v"(lds), "v"(voff), "s"(gbase) : "memory");
}
__device__ __forceinline__ void wait_async0() {
    asm volatile("s_wait_asynccnt 0" ::: "memory");
}

constexpr int kB = 32, kNQ = 300, kD = 256, kH = 8, kHD = 32, kS = 8400, kFFN = 1024;

// ---------------------------------------------------------------------------
// GEMM: C[M,N] = epi( A_bf16[M,K] @ Wt_bf16[N64,K]^T + bias ), N64 = padded N.
// 256 threads = 8 waves; block tile 64x64; wave -> 16x32 of C (2 WMMA accs).
// A rows must cover m0..m0+63 (M multiple of 64); Wt padded to 64-row tiles.
// ---------------------------------------------------------------------------
template <int RELU, int OUTBF>
__global__ __launch_bounds__(256) void gemm_wmma_async(
    const unsigned short* __restrict__ A,
    const unsigned short* __restrict__ Wt,
    const float* __restrict__ bias,
    void* __restrict__ Cv, int M, int N, int K, float scale)
{
    __shared__ __align__(16) unsigned short As[2][64 * 32];
    __shared__ __align__(16) unsigned short Bs[2][64 * 32];

    const int t    = threadIdx.x;
    const int lane = t & 31;
    const int w    = t >> 5;
    const int half = lane >> 4;
    const int ln16 = lane & 15;
    const int m0   = blockIdx.y * 64;
    const int n0   = blockIdx.x * 64;
    const int wm   = (w & 3) * 16;
    const int wn   = (w >> 2) * 32;

    // staging geometry: thread t fills 16B chunk (t&3) of row (t>>2)
    const int srow = t >> 2;
    const int kseg = (t & 3) * 8;                    // bf16 elements
    const unsigned aoff0 = ((unsigned)(m0 + srow) * (unsigned)K + (unsigned)kseg) * 2u;
    const unsigned boff0 = ((unsigned)(n0 + srow) * (unsigned)K + (unsigned)kseg) * 2u;
    const unsigned lrow  = (unsigned)(srow * 64 + (t & 3) * 16);
    unsigned ldsA[2] = { lds_off(&As[0][0]) + lrow, lds_off(&As[1][0]) + lrow };
    unsigned ldsB[2] = { lds_off(&Bs[0][0]) + lrow, lds_off(&Bs[1][0]) + lrow };
    const unsigned long long ga = (unsigned long long)A;
    const unsigned long long gb = (unsigned long long)Wt;

    v8f acc0 = {}; v8f acc1 = {};

    // prologue: fill buffer 0
    async_b128(ldsA[0], ga, aoff0);
    async_b128(ldsB[0], gb, boff0);

    int cur = 0;
    for (int kk = 0; kk < K; kk += 32, cur ^= 1) {
        wait_async0();
        __syncthreads();                             // publish LDS tile
        if (kk + 32 < K) {                           // prefetch next tile
            unsigned ko = (unsigned)(kk + 32) * 2u;
            async_b128(ldsA[cur ^ 1], ga, aoff0 + ko);
            async_b128(ldsB[cur ^ 1], gb, boff0 + ko);
        }
        const unsigned short* as = As[cur];
        const unsigned short* bs = Bs[cur];
        FragBF a, b0, b1;
        #pragma unroll
        for (int j = 0; j < 8; ++j) {
            a.u[j]  = *(const unsigned int*)&as[(wm + ln16) * 32 + kpairA(j, half)];
            int kp  = kpairB(j, half);
            b0.u[j] = *(const unsigned int*)&bs[(wn + ln16) * 32 + kp];
            b1.u[j] = *(const unsigned int*)&bs[(wn + 16 + ln16) * 32 + kp];
        }
        acc0 = __builtin_amdgcn_wmma_f32_16x16x32_bf16(false, a.v, false, b0.v, (short)0, acc0, false, false);
        acc1 = __builtin_amdgcn_wmma_f32_16x16x32_bf16(false, a.v, false, b1.v, (short)0, acc1, false, false);
    }

    #pragma unroll
    for (int r = 0; r < 8; ++r) {
        int m = m0 + wm + r + 8 * half;
        if (m >= M) continue;
        int na = n0 + wn + ln16;
        int nb = na + 16;
        if (na < N) {
            float v = (acc0[r] + bias[na]) * scale;
            if (RELU) v = fmaxf(v, 0.f);
            if (OUTBF) ((unsigned short*)Cv)[(size_t)m * N + na] = f2bf(v);
            else       ((float*)Cv)[(size_t)m * N + na] = v;
        }
        if (nb < N) {
            float v = (acc1[r] + bias[nb]) * scale;
            if (RELU) v = fmaxf(v, 0.f);
            if (OUTBF) ((unsigned short*)Cv)[(size_t)m * N + nb] = f2bf(v);
            else       ((float*)Cv)[(size_t)m * N + nb] = v;
        }
    }
}

// ---------------------------------------------------------------------------
// Flash self-attention, bf16 in / bf16 out. Grid (5, B*H), block 128 (4 waves).
// q pre-scaled by HD^-0.5.
// ---------------------------------------------------------------------------
__global__ __launch_bounds__(128) void attn_wmma(
    const unsigned short* __restrict__ q, const unsigned short* __restrict__ k,
    const unsigned short* __restrict__ v, unsigned short* __restrict__ out)
{
    __shared__ __align__(16) unsigned short Ks[320 * 32];   // [key][hd]
    __shared__ __align__(16) unsigned short Vs[32 * 320];   // [hd][key]
    __shared__ __align__(16) unsigned short Ps[4][16 * 32]; // per-wave probs

    const int bh   = blockIdx.y;
    const int b    = bh >> 3, h = bh & 7;
    const int t    = threadIdx.x;
    const int w    = t >> 5, lane = t & 31;
    const int half = lane >> 4, ln16 = lane & 15;
    const int qt   = blockIdx.x * 4 + w;

    for (int i = 0; i < 80; ++i) {
        int e = t + i * 128;
        int key = e >> 5, hd = e & 31;
        unsigned short kv = 0, vv = 0;
        if (key < kNQ) {
            size_t gi = ((size_t)(b * kNQ + key)) * kD + h * kHD + hd;
            kv = k[gi]; vv = v[gi];
        }
        Ks[key * 32 + hd]  = kv;
        Vs[hd * 320 + key] = vv;
    }
    __syncthreads();

    FragBF qa;
    #pragma unroll
    for (int j = 0; j < 8; ++j) {
        int qrow = qt * 16 + ln16;
        unsigned qv = 0;
        if (qrow < kNQ) {
            size_t gi = ((size_t)(b * kNQ + qrow)) * kD + h * kHD + kpairA(j, half);
            qv = *(const unsigned int*)&q[gi];
        }
        qa.u[j] = qv;
    }

    float rm[8], rl[8];
    #pragma unroll
    for (int r = 0; r < 8; ++r) { rm[r] = -1e30f; rl[r] = 0.f; }
    v8f o0 = {}, o1 = {};

    for (int c = 0; c < 10; ++c) {
        int kb = c * 32;
        FragBF kb0, kb1;
        #pragma unroll
        for (int j = 0; j < 8; ++j) {
            int kp = kpairB(j, half);
            kb0.u[j] = *(const unsigned int*)&Ks[(kb + ln16) * 32 + kp];
            kb1.u[j] = *(const unsigned int*)&Ks[(kb + 16 + ln16) * 32 + kp];
        }
        v8f s0 = {}, s1 = {};
        s0 = __builtin_amdgcn_wmma_f32_16x16x32_bf16(false, qa.v, false, kb0.v, (short)0, s0, false, false);
        s1 = __builtin_amdgcn_wmma_f32_16x16x32_bf16(false, qa.v, false, kb1.v, (short)0, s1, false, false);

        const bool ok0 = (kb + ln16) < kNQ;
        const bool ok1 = (kb + 16 + ln16) < kNQ;
        float cm[8];
        #pragma unroll
        for (int r = 0; r < 8; ++r) {
            float a0 = ok0 ? s0[r] : -1e30f;
            float a1 = ok1 ? s1[r] : -1e30f;
            s0[r] = a0; s1[r] = a1;
            cm[r] = fmaxf(a0, a1);
        }
        #pragma unroll
        for (int o = 1; o < 16; o <<= 1)
            #pragma unroll
            for (int r = 0; r < 8; ++r)
                cm[r] = fmaxf(cm[r], __shfl_xor(cm[r], o, 32));

        float alpha[8], csum[8];
        #pragma unroll
        for (int r = 0; r < 8; ++r) {
            float nm = fmaxf(rm[r], cm[r]);
            alpha[r] = __expf(rm[r] - nm);
            float p0 = __expf(s0[r] - nm);
            float p1 = __expf(s1[r] - nm);
            s0[r] = p0; s1[r] = p1;
            rm[r] = nm;
            csum[r] = p0 + p1;
        }
        #pragma unroll
        for (int o = 1; o < 16; o <<= 1)
            #pragma unroll
            for (int r = 0; r < 8; ++r)
                csum[r] += __shfl_xor(csum[r], o, 32);
        #pragma unroll
        for (int r = 0; r < 8; ++r) {
            rl[r] = rl[r] * alpha[r] + csum[r];
            o0[r] *= alpha[r];
            o1[r] *= alpha[r];
        }

        unsigned short* P = &Ps[w][0];
        #pragma unroll
        for (int r = 0; r < 8; ++r) {
            int m = r + 8 * half;
            P[m * 32 + ln16]      = f2bf(s0[r]);
            P[m * 32 + 16 + ln16] = f2bf(s1[r]);
        }
        __syncthreads();

        FragBF pa, vb0, vb1;
        #pragma unroll
        for (int j = 0; j < 8; ++j) {
            pa.u[j]  = *(const unsigned int*)&P[ln16 * 32 + kpairA(j, half)];
            int kp   = kpairB(j, half);
            vb0.u[j] = *(const unsigned int*)&Vs[ln16 * 320 + kb + kp];
            vb1.u[j] = *(const unsigned int*)&Vs[(16 + ln16) * 320 + kb + kp];
        }
        o0 = __builtin_amdgcn_wmma_f32_16x16x32_bf16(false, pa.v, false, vb0.v, (short)0, o0, false, false);
        o1 = __builtin_amdgcn_wmma_f32_16x16x32_bf16(false, pa.v, false, vb1.v, (short)0, o1, false, false);
        __syncthreads();
    }

    #pragma unroll
    for (int r = 0; r < 8; ++r) {
        int m = qt * 16 + r + 8 * half;
        if (m >= kNQ) continue;
        float inv = 1.f / rl[r];
        size_t base = ((size_t)(b * kNQ + m)) * kD + h * kHD;
        out[base + ln16]      = f2bf(o0[r] * inv);
        out[base + 16 + ln16] = f2bf(o1[r] * inv);
    }
}

// ---------------------------------------------------------------------------
// Deformable sampling (value in bf16, out bf16). Grid B*NQ, 256 thr = (h,hd).
// ---------------------------------------------------------------------------
__global__ __launch_bounds__(256) void deform_sample(
    const unsigned short* __restrict__ value, // [B, S, 256] bf16
    const float* __restrict__ off,            // [B*NQ, 192]
    const float* __restrict__ awl,            // [B*NQ, 96]
    const float* __restrict__ ref,            // [B, NQ, 1, 4]
    unsigned short* __restrict__ out)         // [B*NQ, 256] bf16
{
    const int bq = blockIdx.x;
    const int b  = bq / kNQ;
    const int t  = threadIdx.x;
    const int h  = t >> 5, hd = t & 31;

    float lg[12];
    float mx = -1e30f;
    #pragma unroll
    for (int i = 0; i < 12; ++i) { lg[i] = awl[(size_t)bq * 96 + h * 12 + i]; mx = fmaxf(mx, lg[i]); }
    float sm = 0.f;
    #pragma unroll
    for (int i = 0; i < 12; ++i) { lg[i] = __expf(lg[i] - mx); sm += lg[i]; }
    const float inv = 1.f / sm;

    const float rx = ref[(size_t)bq * 4 + 0], ry = ref[(size_t)bq * 4 + 1];
    const float rw = ref[(size_t)bq * 4 + 2], rh = ref[(size_t)bq * 4 + 3];

    const int lvl_start[3] = {0, 6400, 8000};
    const int lvl_dim[3]   = {80, 40, 20};

    float acc = 0.f;
    for (int l = 0; l < 3; ++l) {
        const int ww = lvl_dim[l], hh = lvl_dim[l], st = lvl_start[l];
        for (int p = 0; p < 4; ++p) {
            int oi = ((h * 3 + l) * 4 + p) * 2;
            float ox = off[(size_t)bq * 192 + oi];
            float oy = off[(size_t)bq * 192 + oi + 1];
            float lx = rx + ox * 0.25f * rw * 0.5f;
            float ly = ry + oy * 0.25f * rh * 0.5f;
            float x = lx * ww - 0.5f;
            float y = ly * hh - 0.5f;
            float x0f = floorf(x), y0f = floorf(y);
            int x0 = (int)x0f, y0 = (int)y0f;
            float wx1 = x - x0f, wy1 = y - y0f;
            float wx0 = 1.f - wx1, wy0 = 1.f - wy1;
            float s = 0.f;
            auto corner = [&](int xi, int yi, float wt) {
                if (xi >= 0 && xi < ww && yi >= 0 && yi < hh) {
                    size_t vi = ((size_t)b * kS + st + yi * ww + xi) * kD + h * kHD + hd;
                    s += wt * bf2f(value[vi]);
                }
            };
            corner(x0,     y0,     wx0 * wy0);
            corner(x0 + 1, y0,     wx1 * wy0);
            corner(x0,     y0 + 1, wx0 * wy1);
            corner(x0 + 1, y0 + 1, wx1 * wy1);
            acc += s * lg[l * 4 + p] * inv;
        }
    }
    out[(size_t)bq * kD + h * kHD + hd] = f2bf(acc);
}

// ---------------------------------------------------------------------------
// out = LN(x + y); optional bf16 copy obf = bf16(out (+ pos)).
// ---------------------------------------------------------------------------
template <int WITHPOS>
__global__ __launch_bounds__(256) void add_ln(
    const float* __restrict__ x, const float* __restrict__ y,
    const float* __restrict__ g, const float* __restrict__ be,
    const float* __restrict__ pos,
    float* __restrict__ out, unsigned short* __restrict__ obf)
{
    __shared__ float red[16];
    const int row = blockIdx.x, t = threadIdx.x;
    const size_t idx = (size_t)row * kD + t;
    float v = x[idx] + y[idx];

    float s = v;
    #pragma unroll
    for (int o = 1; o < 32; o <<= 1) s += __shfl_xor(s, o, 32);
    if ((t & 31) == 0) red[t >> 5] = s;
    __syncthreads();
    float tot = 0.f;
    #pragma unroll
    for (int i = 0; i < 8; ++i) tot += red[i];
    float mean = tot * (1.f / 256.f);

    float d = v - mean;
    float s2 = d * d;
    #pragma unroll
    for (int o = 1; o < 32; o <<= 1) s2 += __shfl_xor(s2, o, 32);
    if ((t & 31) == 0) red[8 + (t >> 5)] = s2;
    __syncthreads();
    float tot2 = 0.f;
    #pragma unroll
    for (int i = 0; i < 8; ++i) tot2 += red[8 + i];
    float var = tot2 * (1.f / 256.f);

    float r = d * rsqrtf(var + 1e-5f) * g[t] + be[t];
    out[idx] = r;
    if (obf) {
        float rb = r;
        if (WITHPOS) rb += pos[idx];
        obf[idx] = f2bf(rb);
    }
}

// ---------------------------------------------------------------------------
// Conversions: fp32 (+fp32) -> bf16 ; weight fp32[K,N] -> bf16[N64,K] (padded)
// ---------------------------------------------------------------------------
__global__ __launch_bounds__(256) void cvt_bf16(
    const float* __restrict__ a, const float* __restrict__ b,
    unsigned short* __restrict__ o, int n)
{
    int i = blockIdx.x * 256 + threadIdx.x;
    if (i < n) {
        float v = a[i];
        if (b) v += b[i];
        o[i] = f2bf(v);
    }
}

__global__ __launch_bounds__(256) void cvt_wt(
    const float* __restrict__ W, unsigned short* __restrict__ Wt,
    int N, int K, int N64)
{
    int i = blockIdx.x * 256 + threadIdx.x;
    if (i < N64 * K) {
        int n = i / K, k = i - n * K;
        float v = (n < N) ? W[(size_t)k * N + n] : 0.f;
        Wt[i] = f2bf(v);
    }
}

// ---------------------------------------------------------------------------
// Orchestration
// ---------------------------------------------------------------------------
static inline dim3 gemm_grid(int M, int N) { return dim3((N + 63) / 64, (M + 63) / 64); }

extern "C" void kernel_launch(void* const* d_in, const int* in_sizes, int n_in,
                              void* d_out, int out_size, void* d_ws, size_t ws_size,
                              hipStream_t stream)
{
    const float* hs   = (const float*)d_in[0];
    const float* pos  = (const float*)d_in[1];
    const float* ref  = (const float*)d_in[2];
    const float* enc  = (const float*)d_in[3];
    const float* Wq   = (const float*)d_in[4];
    const float* Wk   = (const float*)d_in[5];
    const float* Wv   = (const float*)d_in[6];
    const float* Wo   = (const float*)d_in[7];
    const float* Woff = (const float*)d_in[8];
    const float* Waw  = (const float*)d_in[9];
    const float* Wval = (const float*)d_in[10];
    const float* Wout = (const float*)d_in[11];
    const float* W1   = (const float*)d_in[12];
    const float* W2   = (const float*)d_in[13];
    const float* bq   = (const float*)d_in[14];
    const float* bk   = (const float*)d_in[15];
    const float* bv   = (const float*)d_in[16];
    const float* bo   = (const float*)d_in[17];
    const float* boff = (const float*)d_in[18];
    const float* baw  = (const float*)d_in[19];
    const float* bval = (const float*)d_in[20];
    const float* bout = (const float*)d_in[21];
    const float* b1   = (const float*)d_in[22];
    const float* b2   = (const float*)d_in[23];
    const float* ln1g = (const float*)d_in[24];
    const float* ln1b = (const float*)d_in[25];
    const float* ln2g = (const float*)d_in[26];
    const float* ln2b = (const float*)d_in[27];
    const float* ln3g = (const float*)d_in[28];
    const float* ln3b = (const float*)d_in[29];

    const int M  = kB * kNQ;          // 9600
    const int MS = kB * kS;           // 268800

    char* base = (char*)d_ws;
    auto carve = [&](size_t bytes) -> char* {
        char* r = base;
        base += (bytes + 255) & ~(size_t)255;
        return r;
    };
    typedef unsigned short us;
    us* enc_bf   = (us*)carve((size_t)MS * kD * 2);
    us* value_bf = (us*)carve((size_t)MS * kD * 2);
    us* hp1_bf   = (us*)carve((size_t)M * kD * 2);
    us* hs_bf    = (us*)carve((size_t)M * kD * 2);
    us* qbf      = (us*)carve((size_t)M * kD * 2);
    us* kbf      = (us*)carve((size_t)M * kD * 2);
    us* vbf      = (us*)carve((size_t)M * kD * 2);
    us* sa_pre   = (us*)carve((size_t)M * kD * 2);
    us* hp2_bf   = (us*)carve((size_t)M * kD * 2);
    us* ca_pre   = (us*)carve((size_t)M * kD * 2);
    us* hs2_bf   = (us*)carve((size_t)M * kD * 2);
    us* ffn1_bf  = (us*)carve((size_t)M * kFFN * 2);
    float* sa_f   = (float*)carve((size_t)M * kD * 4);
    float* hs1_f  = (float*)carve((size_t)M * kD * 4);
    float* ca_f   = (float*)carve((size_t)M * kD * 4);
    float* hs2_f  = (float*)carve((size_t)M * kD * 4);
    float* ffn2_f = (float*)carve((size_t)M * kD * 4);
    float* offb   = (float*)carve((size_t)M * 192 * 4);
    float* awlb   = (float*)carve((size_t)M * 96 * 4);
    us* Wq_t   = (us*)carve((size_t)256  * 256  * 2);
    us* Wk_t   = (us*)carve((size_t)256  * 256  * 2);
    us* Wv_t   = (us*)carve((size_t)256  * 256  * 2);
    us* Wo_t   = (us*)carve((size_t)256  * 256  * 2);
    us* Woff_t = (us*)carve((size_t)192  * 256  * 2);
    us* Waw_t  = (us*)carve((size_t)128  * 256  * 2);   // N=96 padded to 128
    us* Wval_t = (us*)carve((size_t)256  * 256  * 2);
    us* Wout_t = (us*)carve((size_t)256  * 256  * 2);
    us* W1_t   = (us*)carve((size_t)1024 * 256  * 2);
    us* W2_t   = (us*)carve((size_t)256  * 1024 * 2);

    const float qscale = 0.17677669529663687f;  // 1/sqrt(32)
    auto eg = [](int n) { return dim3((n + 255) / 256); };

    // ---- conversions ----
    cvt_bf16<<<eg(M * kD), 256, 0, stream>>>(hs, pos, hp1_bf, M * kD);
    cvt_bf16<<<eg(M * kD), 256, 0, stream>>>(hs, nullptr, hs_bf, M * kD);
    cvt_bf16<<<eg(MS * kD), 256, 0, stream>>>(enc, nullptr, enc_bf, MS * kD);
    cvt_wt<<<eg(256  * 256), 256, 0, stream>>>(Wq,   Wq_t,   256,  256,  256);
    cvt_wt<<<eg(256  * 256), 256, 0, stream>>>(Wk,   Wk_t,   256,  256,  256);
    cvt_wt<<<eg(256  * 256), 256, 0, stream>>>(Wv,   Wv_t,   256,  256,  256);
    cvt_wt<<<eg(256  * 256), 256, 0, stream>>>(Wo,   Wo_t,   256,  256,  256);
    cvt_wt<<<eg(192  * 256), 256, 0, stream>>>(Woff, Woff_t, 192,  256,  192);
    cvt_wt<<<eg(128  * 256), 256, 0, stream>>>(Waw,  Waw_t,  96,   256,  128);
    cvt_wt<<<eg(256  * 256), 256, 0, stream>>>(Wval, Wval_t, 256,  256,  256);
    cvt_wt<<<eg(256  * 256), 256, 0, stream>>>(Wout, Wout_t, 256,  256,  256);
    cvt_wt<<<eg(1024 * 256), 256, 0, stream>>>(W1,   W1_t,   1024, 256,  1024);
    cvt_wt<<<eg(256  * 1024), 256, 0, stream>>>(W2,  W2_t,   256,  1024, 256);

    // ---- self attention ----
    gemm_wmma_async<0, 1><<<gemm_grid(M, kD), 256, 0, stream>>>(hp1_bf, Wq_t, bq, qbf, M, kD, kD, qscale);
    gemm_wmma_async<0, 1><<<gemm_grid(M, kD), 256, 0, stream>>>(hp1_bf, Wk_t, bk, kbf, M, kD, kD, 1.f);
    gemm_wmma_async<0, 1><<<gemm_grid(M, kD), 256, 0, stream>>>(hs_bf,  Wv_t, bv, vbf, M, kD, kD, 1.f);
    attn_wmma<<<dim3(5, kB * kH), 128, 0, stream>>>(qbf, kbf, vbf, sa_pre);
    gemm_wmma_async<0, 0><<<gemm_grid(M, kD), 256, 0, stream>>>(sa_pre, Wo_t, bo, sa_f, M, kD, kD, 1.f);
    add_ln<1><<<M, 256, 0, stream>>>(hs, sa_f, ln1g, ln1b, pos, hs1_f, hp2_bf);

    // ---- deformable cross attention ----
    gemm_wmma_async<0, 1><<<gemm_grid(MS, kD), 256, 0, stream>>>(enc_bf, Wval_t, bval, value_bf, MS, kD, kD, 1.f);
    gemm_wmma_async<0, 0><<<gemm_grid(M, 192), 256, 0, stream>>>(hp2_bf, Woff_t, boff, offb, M, 192, kD, 1.f);
    gemm_wmma_async<0, 0><<<gemm_grid(M, 96), 256, 0, stream>>>(hp2_bf, Waw_t, baw, awlb, M, 96, kD, 1.f);
    deform_sample<<<M, 256, 0, stream>>>(value_bf, offb, awlb, ref, ca_pre);
    gemm_wmma_async<0, 0><<<gemm_grid(M, kD), 256, 0, stream>>>(ca_pre, Wout_t, bout, ca_f, M, kD, kD, 1.f);
    add_ln<0><<<M, 256, 0, stream>>>(hs1_f, ca_f, ln2g, ln2b, nullptr, hs2_f, hs2_bf);

    // ---- FFN ----
    gemm_wmma_async<1, 1><<<gemm_grid(M, kFFN), 256, 0, stream>>>(hs2_bf, W1_t, b1, ffn1_bf, M, kFFN, kD, 1.f);
    gemm_wmma_async<0, 0><<<gemm_grid(M, kD), 256, 0, stream>>>(ffn1_bf, W2_t, b2, ffn2_f, M, kD, kFFN, 1.f);
    add_ln<0><<<M, 256, 0, stream>>>(hs2_f, ffn2_f, ln3g, ln3b, nullptr, (float*)d_out, nullptr);
}
